// GraphAttentionNetwork_4698694222377
// MI455X (gfx1250) — matrix operand order repaired and checked by reading, not compiled, not run on hardware
//
#include <hip/hip_runtime.h>
#include <hip/hip_bf16.h>

// ---------------------------------------------------------------------------
// GAT (3-layer) for MI455X / gfx1250.
//  - GEMMs on v_wmma_f32_16x16x32_bf16 (bf16 in, f32 accumulate)
//  - Edge softmax/scatter via L2-resident atomics (192MB L2 holds all tables)
// ---------------------------------------------------------------------------

typedef __attribute__((ext_vector_type(16))) __bf16 v16bf;
typedef __attribute__((ext_vector_type(8)))  __bf16 v8bf;
typedef __attribute__((ext_vector_type(8)))  float  v8f;

static __device__ __forceinline__ __bf16 f2bf(float f) {
  unsigned u = __float_as_uint(f);
  unsigned r = (u + 0x7FFFu + ((u >> 16) & 1u)) >> 16;   // RNE
  unsigned short s = (unsigned short)r;
  return __builtin_bit_cast(__bf16, s);
}

// order-preserving float<->uint key for atomicMax-based segment max
static __device__ __forceinline__ unsigned fkey(float f) {
  unsigned u = __float_as_uint(f);
  return u ^ (unsigned)(((int)u >> 31) | 0x80000000);
}
static __device__ __forceinline__ float kval(unsigned k) {
  unsigned u = (k & 0x80000000u) ? (k ^ 0x80000000u) : ~k;
  return __uint_as_float(u);
}

// ---------------------------------------------------------------------------
// f32 -> bf16 elementwise
__global__ void k_f32_to_bf16(const float* __restrict__ in, __bf16* __restrict__ out, int n) {
  int i = blockIdx.x * blockDim.x + threadIdx.x;
  if (i < n) out[i] = f2bf(in[i]);
}

// Repack weight W[K][Nout] (f32 row-major) into WMMA B-fragment order (bf16):
// Bp[ ((tile_n*KT + kt)*32 + lane)*16 + e ] = W[kt*32 + (e/8)*16 + (lane/16)*8 + e%8][tile_n*16 + lane%16]
__global__ void k_pack_w(const float* __restrict__ W, __bf16* __restrict__ Bp, int K, int Nout) {
  int i = blockIdx.x * blockDim.x + threadIdx.x;
  if (i >= K * Nout) return;
  int k = i / Nout, n = i % Nout;
  int kt = k >> 5, kk = k & 31, half = (kk >> 3) & 1, e = (kk & 7) + ((kk >> 4) << 3);
  int tn = n >> 4, col = n & 15, lane = half * 16 + col;
  int KT = K >> 5;
  Bp[((size_t)(tn * KT + kt) * 32 + lane) * 16 + e] = f2bf(W[i]);
}

// ---------------------------------------------------------------------------
// C[NROWS x Nout] (f32) = A[NROWS x K] (bf16, row major) * Bpack (fragment order)
// One wave per 16x16 C tile; wave w of the block handles column tile w.
// grid.x = NROWS/16, blockDim.x = 32*(Nout/16). Exact grid => EXEC all ones.
__global__ void k_gemm_wmma_bf16(const __bf16* __restrict__ A,
                                 const __bf16* __restrict__ Bp,
                                 float* __restrict__ C, int K, int Nout) {
  const int wave = threadIdx.x >> 5;          // = tile_n
  const int lane = threadIdx.x & 31;
  const int half = lane >> 4;
  const int m    = lane & 15;
  const int tile_m = blockIdx.x;
  const int KT = K >> 5;

  const size_t rowBase = (size_t)(tile_m * 16 + m) * (size_t)K;
  v8f acc = {};
#pragma unroll
  for (int kt = 0; kt < 8; ++kt) {            // K = 256 always here
    const int k0 = kt << 5;
    // A fragment: lanes 0-15 rows 0..15 hold K = {k0..k0+7, k0+16..k0+23};
    //             lanes 16-31 hold K = {k0+8..k0+15, k0+24..k0+31}
    v8bf alo = *(const v8bf*)(A + rowBase + k0 + half * 8);
    v8bf ahi = *(const v8bf*)(A + rowBase + k0 + 16 + half * 8);
    v16bf a = __builtin_shufflevector(alo, ahi, 0, 1, 2, 3, 4, 5, 6, 7,
                                      8, 9, 10, 11, 12, 13, 14, 15);
    // B fragment: one contiguous 32B load thanks to the repack
    v16bf b = *(const v16bf*)(Bp + ((size_t)(wave * KT + kt) * 32 + lane) * 16);
    if (kt + 1 < 8) __builtin_prefetch(A + rowBase + k0 + 32, 0, 1);
    acc = __builtin_amdgcn_wmma_f32_16x16x32_bf16(false, a, false, b,
                                                  (short)0, acc, false, false);
  }
  // C layout: element r -> row = r + 8*half (within tile), col = lane%16
  const int col = wave * 16 + m;
#pragma unroll
  for (int r = 0; r < 8; ++r)
    C[(size_t)(tile_m * 16 + half * 8 + r) * (size_t)Nout + col] = acc[r];
}

// ---------------------------------------------------------------------------
// alpha_s[n,h] = sum_c h[n,h,c]*a_src[h,c];  same for a_dst
__global__ void k_alpha(const float* __restrict__ Hf, const float* __restrict__ asrc,
                        const float* __restrict__ adst, float* __restrict__ AS,
                        float* __restrict__ AD, int Nn, int heads, int chid) {
  int i = blockIdx.x * blockDim.x + threadIdx.x;
  if (i >= Nn * heads) return;
  int n = i / heads, h = i % heads;
  const float* hp = Hf + (size_t)n * heads * chid + h * chid;
  const float* as = asrc + h * chid;
  const float* ad = adst + h * chid;
  float s = 0.f, d = 0.f;
  for (int c = 0; c < chid; ++c) { float v = hp[c]; s += v * as[c]; d += v * ad[c]; }
  AS[i] = s; AD[i] = d;
}

// pass 1: e = leaky_relu(as[src]+ad[dst]); EW[gid]=e; segment max via atomicMax
__global__ void k_edge_logits(const float* __restrict__ AS, const float* __restrict__ AD,
                              const int* __restrict__ src, const int* __restrict__ dst,
                              float* __restrict__ EW, unsigned* __restrict__ Mk,
                              int E, int Etot, int heads) {
  int gid = blockIdx.x * blockDim.x + threadIdx.x;
  if (gid >= Etot * heads) return;
  int t = gid / heads, h = gid % heads;
  int s = (t < E) ? src[t] : (t - E);
  int d = (t < E) ? dst[t] : (t - E);
  float e = AS[s * heads + h] + AD[d * heads + h];
  e = (e > 0.f) ? e : 0.2f * e;                // leaky_relu slope 0.2
  EW[gid] = e;
  atomicMax(&Mk[d * heads + h], fkey(e));
}

// pass 2: w = exp(e - m[dst]); EW[gid]=w; denom via atomicAdd
__global__ void k_edge_exp(float* __restrict__ EW, const unsigned* __restrict__ Mk,
                           float* __restrict__ DEN, const int* __restrict__ dst,
                           int E, int Etot, int heads) {
  int gid = blockIdx.x * blockDim.x + threadIdx.x;
  if (gid >= Etot * heads) return;
  int t = gid / heads, h = gid % heads;
  int d = (t < E) ? dst[t] : (t - E);
  float w = __expf(EW[gid] - kval(Mk[d * heads + h]));
  EW[gid] = w;
  atomicAdd(&DEN[d * heads + h], w);
}

// pass 3: out[dst] += h[src] * (w / denom[dst]), float4 per thread
__global__ void k_edge_msg(const float* __restrict__ Hf, const float* __restrict__ EW,
                           const float* __restrict__ DEN, float* __restrict__ OUT,
                           const int* __restrict__ src, const int* __restrict__ dst,
                           int E, int Etot, int heads, int C) {
  int gid = blockIdx.x * blockDim.x + threadIdx.x;
  int Q = C >> 2;
  if (gid >= Etot * Q) return;
  int t = gid / Q, q = gid % Q;
  int s = (t < E) ? src[t] : (t - E);
  int d = (t < E) ? dst[t] : (t - E);
  int head = (q * 4) / (C / heads);
  float alpha = EW[t * heads + head] / (DEN[d * heads + head] + 1e-16f);
  float4 hv = *(const float4*)(Hf + (size_t)s * C + q * 4);
  float* op = OUT + (size_t)d * C + q * 4;
  atomicAdd(op + 0, hv.x * alpha);
  atomicAdd(op + 1, hv.y * alpha);
  atomicAdd(op + 2, hv.z * alpha);
  atomicAdd(op + 3, hv.w * alpha);
}

// epilogue layers 1&2: bias + ELU, emit bf16 activation for the next GEMM
__global__ void k_bias_elu_bf16(const float* __restrict__ OUT, const float* __restrict__ b,
                                __bf16* __restrict__ Abf, int Nn, int C) {
  int i = blockIdx.x * blockDim.x + threadIdx.x;
  if (i >= Nn * C) return;
  float v = OUT[i] + b[i % C];
  v = (v > 0.f) ? v : (__expf(v) - 1.f);       // ELU
  Abf[i] = f2bf(v);
}

// epilogue layer 3: bias only, write node embeddings (f32)
__global__ void k_bias_out(const float* __restrict__ OUT, const float* __restrict__ b,
                           float* __restrict__ node_out, int Nn, int C) {
  int i = blockIdx.x * blockDim.x + threadIdx.x;
  if (i >= Nn * C) return;
  node_out[i] = OUT[i] + b[i % C];
}

__global__ void k_pool_acc(const float* __restrict__ node_out, const int* __restrict__ batch,
                           float* __restrict__ gsum, float* __restrict__ gcnt, int Nn, int C) {
  int i = blockIdx.x * blockDim.x + threadIdx.x;
  if (i >= Nn * C) return;
  int n = i / C, c = i % C;
  int g = batch[n];
  atomicAdd(&gsum[g * C + c], node_out[i]);
  if (c == 0) atomicAdd(&gcnt[g], 1.f);
}

__global__ void k_pool_div(const float* __restrict__ gsum, const float* __restrict__ gcnt,
                           float* __restrict__ gout, int Gg, int C) {
  int i = blockIdx.x * blockDim.x + threadIdx.x;
  if (i >= Gg * C) return;
  gout[i] = gsum[i] / fmaxf(gcnt[i / C], 1.f);
}

// ---------------------------------------------------------------------------
extern "C" void kernel_launch(void* const* d_in, const int* in_sizes, int n_in,
                              void* d_out, int out_size, void* d_ws, size_t ws_size,
                              hipStream_t stream) {
  const int Nn = 50000, Ee = 800000, Gg = 64;
  const int CIN = 256, HID = 64, Hh = 4, OUTC = 64;
  const int Etot = Ee + Nn;

  const float* x    = (const float*)d_in[0];
  const int*  eidx  = (const int*)d_in[1];
  const int*  batch = (const int*)d_in[2];
  const float* W1 = (const float*)d_in[3],  *as1 = (const float*)d_in[4];
  const float* ad1 = (const float*)d_in[5], *b1  = (const float*)d_in[6];
  const float* W2 = (const float*)d_in[7],  *as2 = (const float*)d_in[8];
  const float* ad2 = (const float*)d_in[9], *b2  = (const float*)d_in[10];
  const float* W3 = (const float*)d_in[11], *as3 = (const float*)d_in[12];
  const float* ad3 = (const float*)d_in[13],*b3  = (const float*)d_in[14];
  const int* srcA = eidx;           // edge_index[0]
  const int* dstA = eidx + Ee;      // edge_index[1]

  float* node_out  = (float*)d_out;                 // [Nn*64]
  float* graph_out = (float*)d_out + (size_t)Nn * OUTC;

  // ---- workspace carve-out (~145 MB) ----
  char* ws = (char*)d_ws; size_t off = 0;
  auto take = [&](size_t bytes) -> char* {
    char* p = ws + off; off += (bytes + 255) & ~(size_t)255; return p;
  };
  __bf16* Abf  = (__bf16*)take((size_t)Nn * 256 * 2);
  float*  Hf   = (float*)take((size_t)Nn * 256 * 4);
  float*  OAcc = (float*)take((size_t)Nn * 256 * 4);
  float*  AS   = (float*)take((size_t)Nn * Hh * 4);
  float*  AD   = (float*)take((size_t)Nn * Hh * 4);
  unsigned* Mk = (unsigned*)take((size_t)Nn * Hh * 4);
  float*  DEN  = (float*)take((size_t)Nn * Hh * 4);
  float*  EW   = (float*)take((size_t)Etot * Hh * 4);
  __bf16* W1p  = (__bf16*)take(256 * 256 * 2);
  __bf16* W2p  = (__bf16*)take(256 * 256 * 2);
  __bf16* W3p  = (__bf16*)take(256 * 64 * 2);
  float*  gsum = (float*)take(Gg * OUTC * 4);
  float*  gcnt = (float*)take(Gg * 4);

  const int TB = 256;
  auto nb = [&](long n) { return (int)((n + TB - 1) / TB); };

  // ---- one-time conversions / weight repack ----
  k_f32_to_bf16<<<nb((long)Nn * 256), TB, 0, stream>>>(x, Abf, Nn * 256);
  k_pack_w<<<nb(256 * 256), TB, 0, stream>>>(W1, W1p, 256, 256);
  k_pack_w<<<nb(256 * 256), TB, 0, stream>>>(W2, W2p, 256, 256);
  k_pack_w<<<nb(256 * 64),  TB, 0, stream>>>(W3, W3p, 256, 64);

  auto run_layer = [&](const __bf16* Wp, const float* asrc, const float* adst,
                       int heads, int Cout) {
    const int chid = Cout / heads;
    hipMemsetAsync(OAcc, 0, (size_t)Nn * Cout * 4, stream);
    hipMemsetAsync(Mk,   0, (size_t)Nn * heads * 4, stream);
    hipMemsetAsync(DEN,  0, (size_t)Nn * heads * 4, stream);
    // GEMM: h = A @ W   (exact grid so EXEC==all-ones at every WMMA)
    k_gemm_wmma_bf16<<<Nn / 16, 32 * (Cout / 16), 0, stream>>>(Abf, Wp, Hf, 256, Cout);
    k_alpha<<<nb((long)Nn * heads), TB, 0, stream>>>(Hf, asrc, adst, AS, AD, Nn, heads, chid);
    k_edge_logits<<<nb((long)Etot * heads), TB, 0, stream>>>(AS, AD, srcA, dstA, EW, Mk, Ee, Etot, heads);
    k_edge_exp<<<nb((long)Etot * heads), TB, 0, stream>>>(EW, Mk, DEN, dstA, Ee, Etot, heads);
    k_edge_msg<<<nb((long)Etot * (Cout / 4)), TB, 0, stream>>>(Hf, EW, DEN, OAcc, srcA, dstA,
                                                               Ee, Etot, heads, Cout);
  };

  // layer 1: 256 -> 4x64 (concat), ELU, re-emit bf16
  run_layer(W1p, as1, ad1, Hh, Hh * HID);
  k_bias_elu_bf16<<<nb((long)Nn * 256), TB, 0, stream>>>(OAcc, b1, Abf, Nn, 256);

  // layer 2: 256 -> 4x64 (concat), ELU, re-emit bf16
  run_layer(W2p, as2, ad2, Hh, Hh * HID);
  k_bias_elu_bf16<<<nb((long)Nn * 256), TB, 0, stream>>>(OAcc, b2, Abf, Nn, 256);

  // layer 3: 256 -> 1x64
  run_layer(W3p, as3, ad3, 1, OUTC);
  k_bias_out<<<nb((long)Nn * OUTC), TB, 0, stream>>>(OAcc, b3, node_out, Nn, OUTC);

  // global mean pool
  hipMemsetAsync(gsum, 0, Gg * OUTC * 4, stream);
  hipMemsetAsync(gcnt, 0, Gg * 4, stream);
  k_pool_acc<<<nb((long)Nn * OUTC), TB, 0, stream>>>(node_out, batch, gsum, gcnt, Nn, OUTC);
  k_pool_div<<<nb(Gg * OUTC), TB, 0, stream>>>(gsum, gcnt, graph_out, Gg, OUTC);
}
